// HFMM_86294482911616
// MI455X (gfx1250) — matrix operand order repaired and checked by reading, not compiled, not run on hardware
//
#include <hip/hip_runtime.h>

#define DIMC 64
#define DST  16
#define DIN  128
#define KD   4
#define BSZ  2
#define HH   64
#define WW   64
#define LL   4096
#define NPIX (BSZ*LL)
#define XDC  64   // padded x_proj output columns (36 valid)

typedef __attribute__((ext_vector_type(16))) __bf16          v16bf;
typedef __attribute__((ext_vector_type(16))) unsigned short  v16u;
typedef __attribute__((ext_vector_type(8)))  float           v8f;

static inline int cdiv(int a, int b) { return (a + b - 1) / b; }

__device__ __forceinline__ unsigned short f2bf(float f) {
  union { float f; unsigned u; } v; v.f = f;
  unsigned u = v.u;
  return (unsigned short)((u + 0x7FFFu + ((u >> 16) & 1u)) >> 16);
}
__device__ __forceinline__ float bf2f(unsigned short h) {
  union { unsigned u; float f; } v; v.u = ((unsigned)h) << 16;
  return v.f;
}
__device__ __forceinline__ float siluf(float x) { return x / (1.f + __expf(-x)); }
__device__ __forceinline__ float softplusf(float x) {
  return (x > 20.f) ? x : log1pf(__expf(x));
}

__device__ __forceinline__ v16bf load_fragA(const unsigned short* p) {
  // lane holds row; elements 0..7 = K+0..7 slice, 8..15 = K+16..23 slice (hi-lane shifted)
  v16u u;
#pragma unroll
  for (int e = 0; e < 8; e++) { u[e] = p[e]; u[e + 8] = p[e + 16]; }
  return __builtin_bit_cast(v16bf, u);
}
__device__ __forceinline__ v16bf load_fragB(const unsigned short* p) {
  // lane holds column; elements 0..7 = K+0..7 slice, 8..15 = K+16..23 slice (hi-lane shifted)
  v16u u;
#pragma unroll
  for (int e = 0; e < 8; e++) { u[e] = p[e]; u[e + 8] = p[e + 16]; }
  return __builtin_bit_cast(v16bf, u);
}

// ---------------------------------------------------------------------------
// bf16 WMMA GEMM, 2x2 register tiling: each wave computes a 32x32 block of
// C[M,N] = A[M,K] * B[N,K]^T (+bias[n]).  4 WMMAs per K-slice over 4 fragment
// loads (2 A + 2 B) -> 2x the MACs/byte of the single-tile version, and 4
// independent XDL ops for the scheduler to overlap with VMEM.
// Requires M%32==0, N%32==0, K%32==0.
// ---------------------------------------------------------------------------
__global__ void k_gemm_bf16(const unsigned short* __restrict__ A,
                            const unsigned short* __restrict__ B,
                            const float* __restrict__ bias,
                            float* __restrict__ C,
                            int M, int N, int K) {
  int wid = blockIdx.x * (blockDim.x >> 5) + (threadIdx.x >> 5);
  int tilesN = N >> 5;
  int total = (M >> 5) * tilesN;
  if (wid >= total) return;                 // uniform per wave
  int tm = wid / tilesN;
  int tn = wid - tm * tilesN;
  int lane = threadIdx.x & 31;
  int r16 = lane & 15;
  int hi  = lane >> 4;
  const unsigned short* Arow0 = A + (size_t)(tm * 32 + r16) * K;
  const unsigned short* Arow1 = Arow0 + (size_t)16 * K;
  const unsigned short* Brow0 = B + (size_t)(tn * 32 + r16) * K;
  const unsigned short* Brow1 = Brow0 + (size_t)16 * K;
  v8f acc00 = {0.f,0.f,0.f,0.f,0.f,0.f,0.f,0.f};
  v8f acc01 = acc00, acc10 = acc00, acc11 = acc00;
  for (int kk = 0; kk < K; kk += 32) {
    const unsigned short* a0p = Arow0 + kk + hi * 8;
    const unsigned short* a1p = Arow1 + kk + hi * 8;
    const unsigned short* b0p = Brow0 + kk + hi * 8;
    const unsigned short* b1p = Brow1 + kk + hi * 8;
    if (kk + 32 < K) {                      // prefetch next K slice
      __builtin_prefetch(a0p + 32, 0, 1);
      __builtin_prefetch(a1p + 32, 0, 1);
      __builtin_prefetch(b0p + 32, 0, 1);
      __builtin_prefetch(b1p + 32, 0, 1);
    }
    v16bf a0 = load_fragA(a0p);
    v16bf a1 = load_fragA(a1p);
    v16bf b0 = load_fragB(b0p);
    v16bf b1 = load_fragB(b1p);
    acc00 = __builtin_amdgcn_wmma_f32_16x16x32_bf16(false, a0, false, b0, (short)0, acc00, false, false);
    acc01 = __builtin_amdgcn_wmma_f32_16x16x32_bf16(false, a0, false, b1, (short)0, acc01, false, false);
    acc10 = __builtin_amdgcn_wmma_f32_16x16x32_bf16(false, a1, false, b0, (short)0, acc10, false, false);
    acc11 = __builtin_amdgcn_wmma_f32_16x16x32_bf16(false, a1, false, b1, (short)0, acc11, false, false);
  }
  float bv0 = bias ? bias[tn * 32 + r16] : 0.f;
  float bv1 = bias ? bias[tn * 32 + 16 + r16] : 0.f;
  float* C0 = C + (size_t)(tm * 32 + hi * 8) * N + tn * 32 + r16;
  float* C1 = C + (size_t)(tm * 32 + 16 + hi * 8) * N + tn * 32 + r16;
#pragma unroll
  for (int r = 0; r < 8; r++) {
    C0[(size_t)r * N]      = acc00[r] + bv0;
    C0[(size_t)r * N + 16] = acc01[r] + bv1;
    C1[(size_t)r * N]      = acc10[r] + bv0;
    C1[(size_t)r * N + 16] = acc11[r] + bv1;
  }
}

// ---------------------------------------------------------------------------
// Weight conversion helpers
// ---------------------------------------------------------------------------
__global__ void k_convert_bf16(const float* __restrict__ in,
                               unsigned short* __restrict__ out, int n) {
  int i = blockIdx.x * blockDim.x + threadIdx.x;
  if (i < n) out[i] = f2bf(in[i]);
}

// x_proj_w (K,36,128) -> (K,64,128) zero padded, bf16
__global__ void k_pad_xproj(const float* __restrict__ w,
                            unsigned short* __restrict__ out) {
  int i = blockIdx.x * blockDim.x + threadIdx.x;
  if (i >= KD * XDC * DIN) return;
  int d = i & (DIN - 1);
  int c = (i >> 7) % XDC;
  int k = i / (XDC * DIN);
  out[i] = (c < 36) ? f2bf(w[(k * 36 + c) * DIN + d]) : (unsigned short)0;
}

// ---------------------------------------------------------------------------
// Channel LayerNorm over C=64 of (B,64,H,W) input
// ---------------------------------------------------------------------------
__global__ void k_ln_pm_bf16(const float* __restrict__ x, const float* __restrict__ w,
                             const float* __restrict__ b, unsigned short* __restrict__ out) {
  int p = blockIdx.x * blockDim.x + threadIdx.x;
  if (p >= NPIX) return;
  int bb = p >> 12, l = p & (LL - 1);
  const float* base = x + (size_t)bb * DIMC * LL + l;
  float s = 0.f, s2 = 0.f;
  for (int c = 0; c < DIMC; c++) { float v = base[(size_t)c * LL]; s += v; s2 += v * v; }
  float mu = s * (1.f / DIMC);
  float var = s2 * (1.f / DIMC) - mu * mu;
  float rstd = rsqrtf(var + 1e-5f);
  for (int c = 0; c < DIMC; c++) {
    float v = (base[(size_t)c * LL] - mu) * rstd * w[c] + b[c];
    out[(size_t)p * DIMC + c] = f2bf(v);
  }
}

__global__ void k_ln_cm_f32(const float* __restrict__ x, const float* __restrict__ w,
                            const float* __restrict__ b, float* __restrict__ out) {
  int p = blockIdx.x * blockDim.x + threadIdx.x;
  if (p >= NPIX) return;
  int bb = p >> 12, l = p & (LL - 1);
  const float* base = x + (size_t)bb * DIMC * LL + l;
  float s = 0.f, s2 = 0.f;
  for (int c = 0; c < DIMC; c++) { float v = base[(size_t)c * LL]; s += v; s2 += v * v; }
  float mu = s * (1.f / DIMC);
  float var = s2 * (1.f / DIMC) - mu * mu;
  float rstd = rsqrtf(var + 1e-5f);
  for (int c = 0; c < DIMC; c++) {
    float v = (base[(size_t)c * LL] - mu) * rstd * w[c] + b[c];
    out[((size_t)(bb * DIMC + c)) * LL + l] = v;
  }
}

// ---------------------------------------------------------------------------
// Depthwise 3x3 conv on xx (first 128 cols of pixel-major xxz) + SiLU,
// fanned out to the 4 scan directions as bf16 (b,k,l,d) layout.
// ---------------------------------------------------------------------------
__global__ void k_dwconv_silu_dirs(const float* __restrict__ xxz,
                                   const float* __restrict__ cw,
                                   const float* __restrict__ cb,
                                   unsigned short* __restrict__ xs) {
  int tid = blockIdx.x * blockDim.x + threadIdx.x;
  if (tid >= BSZ * LL * DIN) return;
  int d = tid & (DIN - 1);
  int l = (tid >> 7) & (LL - 1);
  int b = tid >> 19;
  int h = l >> 6, w = l & 63;
  float acc = cb[d];
#pragma unroll
  for (int kh = 0; kh < 3; kh++) {
    int hh = h + kh - 1;
    if (hh < 0 || hh >= HH) continue;
#pragma unroll
    for (int kw = 0; kw < 3; kw++) {
      int ww2 = w + kw - 1;
      if (ww2 < 0 || ww2 >= WW) continue;
      acc += cw[d * 9 + kh * 3 + kw] * xxz[((size_t)(b * LL + hh * WW + ww2)) * 256 + d];
    }
  }
  unsigned short bf = f2bf(siluf(acc));
  int lT = (l & 63) * HH + (l >> 6);
  size_t base = (size_t)b * KD * LL * DIN;
  xs[base + ((size_t)(0 * LL + l)) * DIN + d] = bf;
  xs[base + ((size_t)(1 * LL + lT)) * DIN + d] = bf;
  xs[base + ((size_t)(2 * LL + (LL - 1 - l))) * DIN + d] = bf;
  xs[base + ((size_t)(3 * LL + (LL - 1 - lT))) * DIN + d] = bf;
}

// ---------------------------------------------------------------------------
// Selective scan: one block per (b,k), one lane per d, h[16] in registers.
// Per-step 36 shared scalars staged in LDS; delta computed in-register.
// ---------------------------------------------------------------------------
__global__ void k_scan(const unsigned short* __restrict__ xs,
                       const float* __restrict__ xdbl,
                       const float* __restrict__ dtw,
                       const float* __restrict__ dtb,
                       const float* __restrict__ Alogs,
                       const float* __restrict__ Dsv,
                       unsigned short* __restrict__ outy) {
  int bk = blockIdx.x;
  int d = threadIdx.x;
  int k = bk & 3;
  int kd = k * DIN + d;
  float w0 = dtw[kd * 4 + 0], w1 = dtw[kd * 4 + 1];
  float w2 = dtw[kd * 4 + 2], w3 = dtw[kd * 4 + 3];
  float db = dtb[kd];
  float Dv = Dsv[kd];
  float An[DST];
#pragma unroll
  for (int s = 0; s < DST; s++) An[s] = -__expf(Alogs[kd * DST + s]);
  float hst[DST];
#pragma unroll
  for (int s = 0; s < DST; s++) hst[s] = 0.f;
  __shared__ float sh[36];
  const float* xrow = xdbl + (size_t)bk * LL * XDC;
  const unsigned short* urow = xs + (size_t)bk * LL * DIN;
  unsigned short* yrow = outy + (size_t)bk * LL * DIN;
  for (int t = 0; t < LL; t++) {
    __syncthreads();
    if (d < 36) sh[d] = xrow[(size_t)t * XDC + d];
    __syncthreads();
    float u = bf2f(urow[(size_t)t * DIN + d]);
    float delta = softplusf(db + w0 * sh[0] + w1 * sh[1] + w2 * sh[2] + w3 * sh[3]);
    float du = delta * u;
    float y = 0.f;
#pragma unroll
    for (int s = 0; s < DST; s++) {
      hst[s] = __expf(delta * An[s]) * hst[s] + du * sh[4 + s];
      y += hst[s] * sh[20 + s];
    }
    yrow[(size_t)t * DIN + d] = f2bf(y + u * Dv);
  }
}

// ---------------------------------------------------------------------------
// Merge 4 directions + LayerNorm(128) + SiLU(z) gate -> bf16 GEMM A
// ---------------------------------------------------------------------------
__global__ void k_merge(const unsigned short* __restrict__ outy,
                        const float* __restrict__ xxz,
                        const float* __restrict__ onw,
                        const float* __restrict__ onb,
                        unsigned short* __restrict__ yg) {
  int p = blockIdx.x;
  int d = threadIdx.x;
  int b = p >> 12, l = p & (LL - 1);
  int h = l >> 6, w = l & 63;
  int lT = w * HH + h;
  size_t base = (size_t)b * KD * LL * DIN;
  float v = bf2f(outy[base + ((size_t)(0 * LL + l)) * DIN + d])
          + bf2f(outy[base + ((size_t)(2 * LL + (LL - 1 - l))) * DIN + d])
          + bf2f(outy[base + ((size_t)(1 * LL + lT)) * DIN + d])
          + bf2f(outy[base + ((size_t)(3 * LL + (LL - 1 - lT))) * DIN + d]);
  __shared__ float red[DIN];
  red[d] = v; __syncthreads();
  for (int s = 64; s > 0; s >>= 1) { if (d < s) red[d] += red[d + s]; __syncthreads(); }
  float mu = red[0] * (1.f / DIN);
  __syncthreads();
  float dv = v - mu;
  red[d] = dv * dv; __syncthreads();
  for (int s = 64; s > 0; s >>= 1) { if (d < s) red[d] += red[d + s]; __syncthreads(); }
  float var = red[0] * (1.f / DIN);
  float nv = dv * rsqrtf(var + 1e-5f) * onw[d] + onb[d];
  float z = xxz[((size_t)(b * LL + l)) * 256 + DIN + d];
  yg[((size_t)(b * LL + l)) * DIN + d] = f2bf(nv * siluf(z));
}

// pixel-major [p,Cin] -> channel-major (b, ctot, L) slot [coff..coff+Cin), +optional residual
__global__ void k_pm2cm_add(const float* __restrict__ pm, const float* __restrict__ res,
                            float* __restrict__ out, int Cin, int ctot, int coff) {
  int tid = blockIdx.x * blockDim.x + threadIdx.x;
  if (tid >= BSZ * Cin * LL) return;
  int l = tid & (LL - 1);
  int c = (tid >> 12) % Cin;
  int b = tid / (Cin * LL);
  float v = pm[((size_t)(b * LL + l)) * Cin + c];
  if (res) v += res[((size_t)(b * DIMC + c)) * LL + l];
  out[((size_t)(b * ctot + coff + c)) * LL + l] = v;
}

// depthwise 3x3 pad1, channel-major, optional residual add
__global__ void k_dw3x3(const float* __restrict__ in, const float* __restrict__ w9,
                        const float* __restrict__ bias, const float* __restrict__ res,
                        float* __restrict__ out) {
  int tid = blockIdx.x * blockDim.x + threadIdx.x;
  if (tid >= BSZ * DIMC * LL) return;
  int l = tid & (LL - 1);
  int c = (tid >> 12) & (DIMC - 1);
  int h = l >> 6, w = l & 63;
  const float* src = in + (size_t)(tid - l);
  float acc = bias[c];
#pragma unroll
  for (int kh = 0; kh < 3; kh++) {
    int hh = h + kh - 1;
    if (hh < 0 || hh >= HH) continue;
#pragma unroll
    for (int kw = 0; kw < 3; kw++) {
      int ww2 = w + kw - 1;
      if (ww2 < 0 || ww2 >= WW) continue;
      acc += w9[c * 9 + kh * 3 + kw] * src[hh * WW + ww2];
    }
  }
  out[tid] = (res ? res[tid] : 0.f) + acc;
}

// horizontal separable passes: 1x3, 1x5, 1x7 depthwise
__global__ void k_hpass(const float* __restrict__ t0,
                        const float* __restrict__ w13, const float* __restrict__ b13,
                        const float* __restrict__ w15, const float* __restrict__ b15,
                        const float* __restrict__ w17, const float* __restrict__ b17,
                        float* __restrict__ h1, float* __restrict__ h2, float* __restrict__ h3) {
  int tid = blockIdx.x * blockDim.x + threadIdx.x;
  if (tid >= BSZ * DIMC * LL) return;
  int l = tid & (LL - 1);
  int c = (tid >> 12) & (DIMC - 1);
  int w = l & 63;
  float xv[7];
#pragma unroll
  for (int j = 0; j < 7; j++) {
    int ww2 = w + j - 3;
    xv[j] = (ww2 >= 0 && ww2 < WW) ? t0[tid - w + ww2] : 0.f;
  }
  float a1 = b13[c], a2 = b15[c], a3 = b17[c];
#pragma unroll
  for (int j = 0; j < 3; j++) a1 += w13[c * 3 + j] * xv[2 + j];
#pragma unroll
  for (int j = 0; j < 5; j++) a2 += w15[c * 5 + j] * xv[1 + j];
#pragma unroll
  for (int j = 0; j < 7; j++) a3 += w17[c * 7 + j] * xv[j];
  h1[tid] = a1; h2[tid] = a2; h3[tid] = a3;
}

// vertical passes (3x1, 5x1, 7x1) + identity, packed as bf16 pixel-major cat [p,256]
__global__ void k_vpass_cat(const float* __restrict__ h1, const float* __restrict__ h2,
                            const float* __restrict__ h3, const float* __restrict__ t0,
                            const float* __restrict__ w31, const float* __restrict__ b31,
                            const float* __restrict__ w51, const float* __restrict__ b51,
                            const float* __restrict__ w71, const float* __restrict__ b71,
                            unsigned short* __restrict__ cat) {
  int tid = blockIdx.x * blockDim.x + threadIdx.x;
  if (tid >= BSZ * DIMC * LL) return;
  int l = tid & (LL - 1);
  int c = (tid >> 12) & (DIMC - 1);
  int b = tid >> 18;
  int h = l >> 6;
  float x1 = b31[c], x2 = b51[c], x3 = b71[c];
#pragma unroll
  for (int j = 0; j < 3; j++) {
    int hh = h + j - 1;
    if (hh >= 0 && hh < HH) x1 += w31[c * 3 + j] * h1[tid + (hh - h) * WW];
  }
#pragma unroll
  for (int j = 0; j < 5; j++) {
    int hh = h + j - 2;
    if (hh >= 0 && hh < HH) x2 += w51[c * 5 + j] * h2[tid + (hh - h) * WW];
  }
#pragma unroll
  for (int j = 0; j < 7; j++) {
    int hh = h + j - 3;
    if (hh >= 0 && hh < HH) x3 += w71[c * 7 + j] * h3[tid + (hh - h) * WW];
  }
  size_t pmb = ((size_t)(b * LL + l)) * 256;
  cat[pmb + c]       = f2bf(x1);
  cat[pmb + 64 + c]  = f2bf(x2);
  cat[pmb + 128 + c] = f2bf(x3);
  cat[pmb + 192 + c] = f2bf(t0[tid]);
}

// im2col for 3x3 dilated conv, bf16 out [p, Cin*9] with column = ci*9+kh*3+kw
__global__ void k_im2col3(const float* __restrict__ in, unsigned short* __restrict__ out,
                          int Cin, int ctot, int coff, int dil) {
  int tid = blockIdx.x * blockDim.x + threadIdx.x;
  int total = NPIX * Cin;
  if (tid >= total) return;
  int ci = tid % Cin;
  int p = tid / Cin;
  int b = p >> 12, l = p & (LL - 1);
  int h = l >> 6, w = l & 63;
  const float* src = in + ((size_t)(b * ctot + coff + ci)) * LL;
  unsigned short* dst = out + (size_t)p * (Cin * 9) + ci * 9;
#pragma unroll
  for (int kh = 0; kh < 3; kh++) {
    int hh = h + dil * (kh - 1);
#pragma unroll
    for (int kw = 0; kw < 3; kw++) {
      int ww2 = w + dil * (kw - 1);
      float v = (hh >= 0 && hh < HH && ww2 >= 0 && ww2 < WW) ? src[hh * WW + ww2] : 0.f;
      dst[kh * 3 + kw] = f2bf(v);
    }
  }
}

// ---------------------------------------------------------------------------
extern "C" void kernel_launch(void* const* d_in, const int* in_sizes, int n_in,
                              void* d_out, int out_size, void* d_ws, size_t ws_size,
                              hipStream_t stream) {
  (void)in_sizes; (void)n_in; (void)out_size; (void)ws_size;
  const float* x         = (const float*)d_in[0];
  const float* ln1w      = (const float*)d_in[1];
  const float* ln1b      = (const float*)d_in[2];
  const float* ln2w      = (const float*)d_in[3];
  const float* ln2b      = (const float*)d_in[4];
  const float* in_proj_w = (const float*)d_in[5];
  const float* conv_w    = (const float*)d_in[6];
  const float* conv_b    = (const float*)d_in[7];
  const float* x_proj_w  = (const float*)d_in[8];
  const float* dt_w      = (const float*)d_in[9];
  const float* dt_b      = (const float*)d_in[10];
  const float* A_logs    = (const float*)d_in[11];
  const float* Ds        = (const float*)d_in[12];
  const float* onw       = (const float*)d_in[13];
  const float* onb       = (const float*)d_in[14];
  const float* out_proj_w= (const float*)d_in[15];
  const float* init_w    = (const float*)d_in[16];
  const float* init_b    = (const float*)d_in[17];
  const float* w13 = (const float*)d_in[18]; const float* b13 = (const float*)d_in[19];
  const float* w31 = (const float*)d_in[20]; const float* b31 = (const float*)d_in[21];
  const float* w15 = (const float*)d_in[22]; const float* b15 = (const float*)d_in[23];
  const float* w51 = (const float*)d_in[24]; const float* b51 = (const float*)d_in[25];
  const float* w17 = (const float*)d_in[26]; const float* b17 = (const float*)d_in[27];
  const float* w71 = (const float*)d_in[28]; const float* b71 = (const float*)d_in[29];
  const float* mix_w = (const float*)d_in[30]; const float* mix_b = (const float*)d_in[31];
  const float* d1_w  = (const float*)d_in[32]; const float* d1_b  = (const float*)d_in[33];
  const float* d2_w  = (const float*)d_in[34]; const float* d2_b  = (const float*)d_in[35];
  const float* d3_w  = (const float*)d_in[36]; const float* d3_b  = (const float*)d_in[37];
  const float* out_w = (const float*)d_in[38]; const float* out_b = (const float*)d_in[39];

  char* wsb = (char*)d_ws;
  size_t off = 0;
  auto alloc = [&](size_t bytes) -> char* {
    char* p = wsb + off;
    off += (bytes + 255) & ~(size_t)255;
    return p;
  };

  unsigned short* xln   = (unsigned short*)alloc((size_t)NPIX * DIMC * 2);
  unsigned short* wb_in = (unsigned short*)alloc((size_t)16384 * 2);
  unsigned short* wb_xp = (unsigned short*)alloc((size_t)KD * XDC * DIN * 2);
  unsigned short* wb_op = (unsigned short*)alloc((size_t)8192 * 2);
  unsigned short* wb_mx = (unsigned short*)alloc((size_t)16384 * 2);
  unsigned short* wb_d1 = (unsigned short*)alloc((size_t)18432 * 2);
  unsigned short* wb_d2 = (unsigned short*)alloc((size_t)9216 * 2);
  unsigned short* wb_d3 = (unsigned short*)alloc((size_t)36864 * 2);
  float*          xxz   = (float*)alloc((size_t)NPIX * 256 * 4);
  unsigned short* xsb   = (unsigned short*)alloc((size_t)BSZ * KD * LL * DIN * 2);
  float*          xdbl  = (float*)alloc((size_t)BSZ * KD * LL * XDC * 4);
  unsigned short* outy  = (unsigned short*)alloc((size_t)BSZ * KD * LL * DIN * 2);
  unsigned short* yg    = (unsigned short*)alloc((size_t)NPIX * DIN * 2);
  float*          pm    = (float*)alloc((size_t)NPIX * 64 * 4);
  float*          yres  = (float*)alloc((size_t)BSZ * DIMC * LL * 4);
  float*          ln2o  = (float*)alloc((size_t)BSZ * DIMC * LL * 4);
  float*          t0    = (float*)alloc((size_t)BSZ * DIMC * LL * 4);
  float*          hb1   = (float*)alloc((size_t)BSZ * DIMC * LL * 4);
  float*          hb2   = (float*)alloc((size_t)BSZ * DIMC * LL * 4);
  float*          hb3   = (float*)alloc((size_t)BSZ * DIMC * LL * 4);
  unsigned short* cat   = (unsigned short*)alloc((size_t)NPIX * 256 * 2);
  float*          xmix  = (float*)alloc((size_t)BSZ * DIMC * LL * 4);
  float*          d12   = (float*)alloc((size_t)BSZ * DIMC * LL * 4);
  unsigned short* im2c  = (unsigned short*)alloc((size_t)NPIX * 576 * 2);
  float*          t3    = (float*)alloc((size_t)BSZ * DIMC * LL * 4);

  const int TB = 256;
  auto gemm = [&](const unsigned short* A, const unsigned short* B, const float* bias,
                  float* C, int M, int N, int K) {
    int tiles = (M / 32) * (N / 32);        // one 32x32 block per wave
    k_gemm_bf16<<<cdiv(tiles, 8), 256, 0, stream>>>(A, B, bias, C, M, N, K);
  };

  // weight conversion
  k_convert_bf16<<<cdiv(16384, TB), TB, 0, stream>>>(in_proj_w, wb_in, 16384);
  k_pad_xproj<<<cdiv(KD * XDC * DIN, TB), TB, 0, stream>>>(x_proj_w, wb_xp);
  k_convert_bf16<<<cdiv(8192, TB), TB, 0, stream>>>(out_proj_w, wb_op, 8192);
  k_convert_bf16<<<cdiv(16384, TB), TB, 0, stream>>>(mix_w, wb_mx, 16384);
  k_convert_bf16<<<cdiv(18432, TB), TB, 0, stream>>>(d1_w, wb_d1, 18432);
  k_convert_bf16<<<cdiv(9216, TB), TB, 0, stream>>>(d2_w, wb_d2, 9216);
  k_convert_bf16<<<cdiv(36864, TB), TB, 0, stream>>>(d3_w, wb_d3, 36864);

  // ---- VSSM branch ----
  k_ln_pm_bf16<<<cdiv(NPIX, TB), TB, 0, stream>>>(x, ln1w, ln1b, xln);
  gemm(xln, wb_in, nullptr, xxz, NPIX, 256, DIMC);                 // in_proj
  k_dwconv_silu_dirs<<<cdiv(BSZ * LL * DIN, TB), TB, 0, stream>>>(xxz, conv_w, conv_b, xsb);
  for (int b = 0; b < BSZ; b++)
    for (int k = 0; k < KD; k++)                                    // x_proj per (b,k)
      gemm(xsb + ((size_t)(b * KD + k)) * LL * DIN,
           wb_xp + (size_t)k * XDC * DIN, nullptr,
           xdbl + ((size_t)(b * KD + k)) * LL * XDC, LL, XDC, DIN);
  k_scan<<<BSZ * KD, DIN, 0, stream>>>(xsb, xdbl, dt_w, dt_b, A_logs, Ds, outy);
  k_merge<<<NPIX, DIN, 0, stream>>>(outy, xxz, onw, onb, yg);
  gemm(yg, wb_op, nullptr, pm, NPIX, DIMC, DIN);                   // out_proj
  k_pm2cm_add<<<cdiv(BSZ * DIMC * LL, TB), TB, 0, stream>>>(pm, x, yres, DIMC, DIMC, 0);

  // ---- LKCM branch ----
  k_ln_cm_f32<<<cdiv(NPIX, TB), TB, 0, stream>>>(yres, ln2w, ln2b, ln2o);
  k_dw3x3<<<cdiv(BSZ * DIMC * LL, TB), TB, 0, stream>>>(ln2o, init_w, init_b, nullptr, t0);
  k_hpass<<<cdiv(BSZ * DIMC * LL, TB), TB, 0, stream>>>(t0, w13, b13, w15, b15, w17, b17,
                                                        hb1, hb2, hb3);
  k_vpass_cat<<<cdiv(BSZ * DIMC * LL, TB), TB, 0, stream>>>(hb1, hb2, hb3, t0,
                                                            w31, b31, w51, b51, w71, b71, cat);
  gemm(cat, wb_mx, mix_b, pm, NPIX, DIMC, 256);                    // mix 1x1
  k_pm2cm_add<<<cdiv(BSZ * DIMC * LL, TB), TB, 0, stream>>>(pm, nullptr, xmix, DIMC, DIMC, 0);

  k_im2col3<<<cdiv(NPIX * 64, TB), TB, 0, stream>>>(xmix, im2c, 64, 64, 0, 2);
  gemm(im2c, wb_d1, d1_b, pm, NPIX, 32, 576);                      // d1
  k_pm2cm_add<<<cdiv(BSZ * 32 * LL, TB), TB, 0, stream>>>(pm, nullptr, d12, 32, 64, 0);

  k_im2col3<<<cdiv(NPIX * 32, TB), TB, 0, stream>>>(d12, im2c, 32, 64, 0, 4);
  gemm(im2c, wb_d2, d2_b, pm, NPIX, 32, 288);                      // d2
  k_pm2cm_add<<<cdiv(BSZ * 32 * LL, TB), TB, 0, stream>>>(pm, nullptr, d12, 32, 64, 32);

  k_im2col3<<<cdiv(NPIX * 64, TB), TB, 0, stream>>>(d12, im2c, 64, 64, 0, 2);
  gemm(im2c, wb_d3, d3_b, pm, NPIX, DIMC, 576);                    // d3
  k_pm2cm_add<<<cdiv(BSZ * DIMC * LL, TB), TB, 0, stream>>>(pm, xmix, t3, DIMC, DIMC, 0);

  // final depthwise + overall residual -> d_out (B,64,H,W)
  k_dw3x3<<<cdiv(BSZ * DIMC * LL, TB), TB, 0, stream>>>(t3, out_w, out_b, yres, (float*)d_out);
}
